// YadGNN_10445360464235
// MI455X (gfx1250) — compile-verified, matrix operand order checked by reading, not run on hardware
//
#include <hip/hip_runtime.h>
#include <hip/hip_bf16.h>
#include <math.h>

typedef _Float16 h16;
typedef __attribute__((ext_vector_type(16))) _Float16 v16h;
typedef __attribute__((ext_vector_type(8)))  _Float16 v8h;
typedef __attribute__((ext_vector_type(8)))  float    v8f;
typedef __attribute__((ext_vector_type(4)))  float    v4f;

// ---------------------------------------------------------------- utilities
__global__ void k_zero(float* p, int n) {
  int i = blockIdx.x * blockDim.x + threadIdx.x;
  if (i < n) p[i] = 0.f;
}

// float atomic-max via ordered-int trick (EMAX initialized to -inf)
__device__ inline void atomicMaxF(float* addr, float val) {
  int vi = __float_as_int(val);
  if (vi >= 0) atomicMax((int*)addr, vi);
  else         atomicMin((unsigned int*)addr, (unsigned int)vi);
}

// ---------------------------------------------------------------- B swizzle
// Convert f32 weight [K, Nc] row-major (rows >= Kvalid read as 0) into the WMMA
// B-fragment-ready f16 layout: D[((kt*(Nc/16)+ct)*32 + lane)*16 + i] holds the
// i-th half of lane `lane`'s fragment for K-tile kt, 16-col tile ct.
// Fragment layout per ISA: lanes 0-15 hold K 0-7/16-23, lanes 16-31 K 8-15/24-31.
__global__ void k_cvt_swz(const float* __restrict__ S, h16* __restrict__ D,
                          int K, int Nc, int Kvalid) {
  int idx = blockIdx.x * blockDim.x + threadIdx.x;
  if (idx >= K * Nc) return;
  int i    = idx & 15;
  int lane = (idx >> 4) & 31;
  int tile = idx >> 9;                 // kt*(Nc/16) + ct
  int ctiles = Nc >> 4;
  int kt = tile / ctiles, ct = tile % ctiles;
  int l15 = lane & 15, hi = lane >> 4;
  int krow = kt * 32 + hi * 8 + ((i < 8) ? i : (8 + i));  // i>=8 -> +16+(i-8)
  int col  = ct * 16 + l15;
  float v = (krow < Kvalid) ? S[(size_t)krow * Nc + col] : 0.f;
  D[idx] = (h16)v;
}

// ------------------------------------------------------- embedding -> f16 A
__global__ void k_embed(const int* __restrict__ xcat,
                        const float* __restrict__ ewid, const float* __restrict__ eken,
                        const float* __restrict__ elrg, const float* __restrict__ esml,
                        h16* __restrict__ G, int Nn) {
  int i = blockIdx.x * blockDim.x + threadIdx.x;
  if (i >= Nn * 96) return;
  int n = i / 96, c = i % 96;
  const int* xc = xcat + n * 4;
  float v;
  if (c < 16)      v = ewid[xc[0] * 16 + c];
  else if (c < 32) v = eken[xc[1] * 16 + (c - 16)];
  else if (c < 64) v = elrg[xc[2] * 32 + (c - 32)];
  else             v = esml[xc[3] * 32 + (c - 64)];
  G[i] = (h16)v;
}

// ---------------------------------------------- WMMA f16 GEMM  C = A@B (+bias)
// A: [M,K] f16 row-major. Bs: swizzled fragment layout (k_cvt_swz).
// One wave computes a 16x64 tile: one A fragment feeds 4 v_wmma per K-step.
// Requires M%16==0, Nc%64==0, K%32==0.
__global__ void k_wmma_gemm(const h16* __restrict__ A, const h16* __restrict__ Bs,
                            const float* __restrict__ bias, float* __restrict__ C,
                            int M, int Nc, int K) {
  int wave = blockIdx.x * (blockDim.x >> 5) + (threadIdx.x >> 5);
  int lane = threadIdx.x & 31;
  int nt = Nc >> 6;                      // 64-col tiles
  int tiles = (M >> 4) * nt;
  if (wave >= tiles) return;
  int tm = (wave / nt) << 4;
  int tn = (wave % nt) << 6;
  int l15 = lane & 15;
  int hi  = lane >> 4;
  int ctiles = Nc >> 4;                  // 16-col tiles across B
  int ct0 = tn >> 4;
  v8f acc0 = {}, acc1 = {}, acc2 = {}, acc3 = {};
  for (int k0 = 0; k0 < K; k0 += 32) {
    // A fragment (two 16B contiguous chunks per lane)
    const h16* ap = A + (size_t)(tm + l15) * K + (k0 + hi * 8);
    v8h alo = *(const v8h*)ap;
    v8h ahi = *(const v8h*)(ap + 16);
    v16h a = __builtin_shufflevector(alo, ahi,
                                     0, 1, 2, 3, 4, 5, 6, 7,
                                     8, 9, 10, 11, 12, 13, 14, 15);
    __builtin_prefetch((const void*)(ap + 32), 0, 3);
    // B fragments: one 32B contiguous vector load each (pre-swizzled)
    const h16* bbase = Bs + (((size_t)(k0 >> 5) * ctiles + ct0) * 32 + lane) * 16;
    v16h b0 = *(const v16h*)(bbase);
    v16h b1 = *(const v16h*)(bbase + 32 * 16);
    v16h b2 = *(const v16h*)(bbase + 2 * 32 * 16);
    v16h b3 = *(const v16h*)(bbase + 3 * 32 * 16);
    acc0 = __builtin_amdgcn_wmma_f32_16x16x32_f16(false, a, false, b0, (short)0, acc0, false, false);
    acc1 = __builtin_amdgcn_wmma_f32_16x16x32_f16(false, a, false, b1, (short)0, acc1, false, false);
    acc2 = __builtin_amdgcn_wmma_f32_16x16x32_f16(false, a, false, b2, (short)0, acc2, false, false);
    acc3 = __builtin_amdgcn_wmma_f32_16x16x32_f16(false, a, false, b3, (short)0, acc3, false, false);
  }
  // C/D layout: col = tile_col + (lane&15), row = tm + 8*(lane>=16) + v
  int rbase = tm + hi * 8;
  v8f* accs[4] = {&acc0, &acc1, &acc2, &acc3};
#pragma unroll
  for (int t = 0; t < 4; ++t) {
    int col = tn + t * 16 + l15;
    float bv = bias ? bias[col] : 0.f;
    v8f av = *accs[t];
#pragma unroll
    for (int v = 0; v < 8; ++v)
      C[(size_t)(rbase + v) * Nc + col] = av[v] + bv;
  }
}

// -------------------------------------------- build padded [N,288] f16 input
__global__ void k_build_a2(const float* __restrict__ CD, const float* __restrict__ xnum,
                           h16* __restrict__ A2, int Nn) {
  int i = blockIdx.x * blockDim.x + threadIdx.x;
  if (i >= Nn * 288) return;
  int n = i / 288, k = i % 288;
  float v = 0.f;
  if (k < 256)      v = CD[n * 256 + k];
  else if (k < 272) v = xnum[n * 16 + (k - 256)];
  A2[i] = (h16)v;
}

// ---------------------------------------------- graph LayerNorm (global stats)
__global__ void k_ln_reduce(const float* __restrict__ X, float* __restrict__ stats, int tot) {
  int i = blockIdx.x * blockDim.x + threadIdx.x;
  float v = (i < tot) ? X[i] : 0.f;
  float s1 = v, s2 = v * v;
  for (int off = 16; off; off >>= 1) {
    s1 += __shfl_xor(s1, off, 32);
    s2 += __shfl_xor(s2, off, 32);
  }
  __shared__ float sh1[8], sh2[8];
  int w = threadIdx.x >> 5;
  if ((threadIdx.x & 31) == 0) { sh1[w] = s1; sh2[w] = s2; }
  __syncthreads();
  if (threadIdx.x == 0) {
    float t1 = 0.f, t2 = 0.f;
    for (int j = 0; j < 8; ++j) { t1 += sh1[j]; t2 += sh2[j]; }
    atomicAdd(&stats[0], t1);
    atomicAdd(&stats[1], t2);
  }
}

__global__ void k_ln_norm_relu(const float* __restrict__ X, const float* __restrict__ stats,
                               const float* __restrict__ w, const float* __restrict__ b,
                               h16* __restrict__ H, int tot) {
  int i = blockIdx.x * blockDim.x + threadIdx.x;
  if (i >= tot) return;
  float M   = (float)tot;
  float mu  = stats[0] / M;
  float var = stats[1] / M - mu * mu;
  float scl = 1.f / (sqrtf(fmaxf(var, 0.f)) + 1e-5f);
  int c = i & 255;
  float h = (X[i] - mu) * scl * w[c] + b[c];
  H[i] = (h16)fmaxf(h, 0.f);
}

// ------------------------------------- x_new = x_prev + concat(bias_f, bias_r)
__global__ void k_init_layer(float* __restrict__ XN, const float* __restrict__ X,
                             const float* __restrict__ bf, const float* __restrict__ br, int Nn) {
  int i = blockIdx.x * blockDim.x + threadIdx.x;
  if (i >= Nn * 256) return;
  int c = i & 255;
  XN[i] = X[i] + ((c < 128) ? bf[c] : br[c - 128]);
}

__global__ void k_init_dir(float* __restrict__ EMX, float* __restrict__ DEN, int Nn) {
  int i = blockIdx.x * blockDim.x + threadIdx.x;
  if (i < Nn) { EMX[i] = -INFINITY; DEN[i] = 0.f; }
}

// -------------------------------- edge attention logits (one wave32 per edge)
__global__ void k_edge_logits(const int* __restrict__ src, const int* __restrict__ dst,
                              const float* __restrict__ eattr, const float* __restrict__ We,
                              const float* __restrict__ att,
                              const float* __restrict__ XL, const float* __restrict__ XR,
                              float* __restrict__ EB, float* __restrict__ EMX, int Ee) {
  int wave = blockIdx.x * (blockDim.x >> 5) + (threadIdx.x >> 5);
  int lane = threadIdx.x & 31;
  if (wave >= Ee) return;
  int s = src[wave], d = dst[wave];
  v4f ea0 = *(const v4f*)(eattr + (size_t)wave * 8);
  v4f ea1 = *(const v4f*)(eattr + (size_t)wave * 8 + 4);
  // ea[c] = sum_k edge_attr[e,k] * We[k,c]  for c = lane*4 .. lane*4+3
  v4f ea = {};
#pragma unroll
  for (int k = 0; k < 4; ++k)
    ea += ea0[k] * (*(const v4f*)(We + k * 128 + lane * 4));
#pragma unroll
  for (int k = 0; k < 4; ++k)
    ea += ea1[k] * (*(const v4f*)(We + (4 + k) * 128 + lane * 4));
  v4f xl = *(const v4f*)(XL + (size_t)s * 128 + lane * 4);
  v4f xr = *(const v4f*)(XR + (size_t)d * 128 + lane * 4);
  v4f at = *(const v4f*)(att + lane * 4);
  float part = 0.f;
#pragma unroll
  for (int j = 0; j < 4; ++j) {
    float m = xl[j] + xr[j] + ea[j];
    m = (m > 0.f) ? m : 0.2f * m;          // leaky_relu(0.2)
    part += m * at[j];
  }
  for (int off = 16; off; off >>= 1) part += __shfl_xor(part, off, 32);
  if (lane == 0) { EB[wave] = part; atomicMaxF(&EMX[d], part); }
}

__global__ void k_edge_exp(float* __restrict__ EB, const int* __restrict__ dst,
                           const float* __restrict__ EMX, float* __restrict__ DEN, int Ee) {
  int i = blockIdx.x * blockDim.x + threadIdx.x;
  if (i >= Ee) return;
  int d = dst[i];
  float em = EMX[d];
  if (!(em > -1e37f)) em = 0.f;            // isfinite fix (init was -inf)
  float ex = expf(EB[i] - em);
  EB[i] = ex;
  atomicAdd(&DEN[d], ex);
}

__global__ void k_edge_scatter(const int* __restrict__ src, const int* __restrict__ dst,
                               const float* __restrict__ EB, const float* __restrict__ DEN,
                               const float* __restrict__ XL, float* __restrict__ XN,
                               int colOff, int Ee) {
  int wave = blockIdx.x * (blockDim.x >> 5) + (threadIdx.x >> 5);
  int lane = threadIdx.x & 31;
  if (wave >= Ee) return;
  int s = src[wave], d = dst[wave];
  float alpha = EB[wave] / (DEN[d] + 1e-16f);
  v4f xs = *(const v4f*)(XL + (size_t)s * 128 + lane * 4);
  float* xo = XN + (size_t)d * 256 + colOff + lane * 4;
#pragma unroll
  for (int j = 0; j < 4; ++j) atomicAdd(&xo[j], xs[j] * alpha);
}

// ------------------------------------------------ out[n] = x[n].W2 + b2
__global__ void k_final(const float* __restrict__ X, const float* __restrict__ W2,
                        const float* __restrict__ b2, float* __restrict__ out, int Nn) {
  int wave = blockIdx.x * (blockDim.x >> 5) + (threadIdx.x >> 5);
  int lane = threadIdx.x & 31;
  if (wave >= Nn) return;
  v4f x0 = *(const v4f*)(X + (size_t)wave * 256 + lane * 8);
  v4f x1 = *(const v4f*)(X + (size_t)wave * 256 + lane * 8 + 4);
  v4f w0 = *(const v4f*)(W2 + lane * 8);
  v4f w1 = *(const v4f*)(W2 + lane * 8 + 4);
  float p = 0.f;
#pragma unroll
  for (int j = 0; j < 4; ++j) p += x0[j] * w0[j] + x1[j] * w1[j];
  for (int off = 16; off; off >>= 1) p += __shfl_xor(p, off, 32);
  if (lane == 0) out[wave] = p + b2[0];
}

// ================================================================ launcher
extern "C" void kernel_launch(void* const* d_in, const int* in_sizes, int n_in,
                              void* d_out, int out_size, void* d_ws, size_t ws_size,
                              hipStream_t stream) {
  // setup_inputs() order: x_cat, x_num, edge_index, edge_attr, then params leaves
  const int*   x_cat  = (const int*)  d_in[0];
  const float* x_num  = (const float*)d_in[1];
  const int*   eidx   = (const int*)  d_in[2];
  const float* eattr  = (const float*)d_in[3];
  const float* emb_wid= (const float*)d_in[4];
  const float* emb_ken= (const float*)d_in[5];
  const float* emb_lrg= (const float*)d_in[6];
  const float* emb_sml= (const float*)d_in[7];
  const float* cd_W   = (const float*)d_in[8];
  const float* cd_b   = (const float*)d_in[9];
  const float* W1     = (const float*)d_in[10];
  const float* b1     = (const float*)d_in[11];
  const float* Wl_f   = (const float*)d_in[12];
  const float* Wr_f   = (const float*)d_in[13];
  const float* We_f   = (const float*)d_in[14];
  const float* att_f  = (const float*)d_in[15];
  const float* bias_f = (const float*)d_in[16];
  const float* Wl_r   = (const float*)d_in[17];
  const float* Wr_r   = (const float*)d_in[18];
  const float* We_r   = (const float*)d_in[19];
  const float* att_r  = (const float*)d_in[20];
  const float* bias_r = (const float*)d_in[21];
  const float* ln_w   = (const float*)d_in[22];
  const float* ln_b   = (const float*)d_in[23];
  const float* W2     = (const float*)d_in[24];
  const float* b2     = (const float*)d_in[25];

  const int Nn = in_sizes[1] / 16;   // x_num [N,16]
  const int Ee = in_sizes[3] / 8;    // edge_attr [E,8]
  const int* src = eidx;             // edge_index [2,E]
  const int* dst = eidx + Ee;

  // ------- carve workspace (256B aligned regions)
  char* wp = (char*)d_ws;
  auto carve = [&](size_t bytes) -> char* {
    char* p = wp; wp += (bytes + 255) & ~(size_t)255; return p;
  };
  float* X    = (float*)carve((size_t)Nn * 256 * 4);
  float* XN   = (float*)carve((size_t)Nn * 256 * 4);
  float* XL   = (float*)carve((size_t)Nn * 128 * 4);
  float* XR   = (float*)carve((size_t)Nn * 128 * 4);
  h16*   H16  = (h16*)  carve((size_t)Nn * 256 * 2);
  h16*   A16  = (h16*)  carve((size_t)Nn * 288 * 2);  // G16 [N,96] then A2 [N,288]
  float* EB   = (float*)carve((size_t)Ee * 4);
  float* EMX  = (float*)carve((size_t)Nn * 4);
  float* DEN  = (float*)carve((size_t)Nn * 4);
  float* STATS= (float*)carve(256);
  h16* cdW16  = (h16*)carve((size_t)96 * 256 * 2);    // swizzled
  h16* W1p16  = (h16*)carve((size_t)288 * 256 * 2);   // swizzled, K padded
  h16* WL16[2][3]; h16* WR16[2][3];
  for (int d = 0; d < 2; ++d)
    for (int i = 0; i < 3; ++i) {
      WL16[d][i] = (h16*)carve((size_t)256 * 128 * 2);
      WR16[d][i] = (h16*)carve((size_t)256 * 128 * 2);
    }

  auto swz = [&](const float* s, h16* d, int K, int Nc, int Kvalid) {
    int total = K * Nc;
    k_cvt_swz<<<(total + 255) / 256, 256, 0, stream>>>(s, d, K, Nc, Kvalid);
  };

  // ------- weight conversion to fragment-swizzled f16 (pad W1 K: 272 -> 288)
  swz(cd_W, cdW16, 96, 256, 96);
  swz(W1,   W1p16, 288, 256, 272);
  for (int i = 0; i < 3; ++i) {
    swz(Wl_f + (size_t)i * 32768, WL16[0][i], 256, 128, 256);
    swz(Wr_f + (size_t)i * 32768, WR16[0][i], 256, 128, 256);
    swz(Wl_r + (size_t)i * 32768, WL16[1][i], 256, 128, 256);
    swz(Wr_r + (size_t)i * 32768, WR16[1][i], 256, 128, 256);
  }

  // ------- embedder + input MLP (WMMA GEMMs)
  k_embed<<<(Nn * 96 + 255) / 256, 256, 0, stream>>>(x_cat, emb_wid, emb_ken,
                                                     emb_lrg, emb_sml, A16, Nn);
  {
    int waves = (Nn / 16) * (256 / 64);
    k_wmma_gemm<<<(waves + 3) / 4, 128, 0, stream>>>(A16, cdW16, cd_b, XN, Nn, 256, 96);
  }
  k_build_a2<<<(Nn * 288 + 255) / 256, 256, 0, stream>>>(XN, x_num, A16, Nn);
  {
    int waves = (Nn / 16) * (256 / 64);
    k_wmma_gemm<<<(waves + 3) / 4, 128, 0, stream>>>(A16, W1p16, b1, X, Nn, 256, 288);
  }

  // ------- GATv2 layers
  const float* WeD[2]  = {We_f, We_r};
  const float* attD[2] = {att_f, att_r};
  float* Xc = X;
  float* Xn = XN;
  const int tot = Nn * 256;
  for (int i = 0; i < 3; ++i) {
    k_zero<<<1, 32, 0, stream>>>(STATS, 2);
    k_ln_reduce<<<(tot + 255) / 256, 256, 0, stream>>>(Xc, STATS, tot);
    k_ln_norm_relu<<<(tot + 255) / 256, 256, 0, stream>>>(Xc, STATS,
        ln_w + (size_t)i * 256, ln_b + (size_t)i * 256, H16, tot);
    k_init_layer<<<(tot + 255) / 256, 256, 0, stream>>>(Xn, Xc,
        bias_f + (size_t)i * 128, bias_r + (size_t)i * 128, Nn);
    for (int d = 0; d < 2; ++d) {
      int waves = (Nn / 16) * (128 / 64);
      k_wmma_gemm<<<(waves + 3) / 4, 128, 0, stream>>>(H16, WL16[d][i], nullptr, XL, Nn, 128, 256);
      k_wmma_gemm<<<(waves + 3) / 4, 128, 0, stream>>>(H16, WR16[d][i], nullptr, XR, Nn, 128, 256);
      const int* sD = (d == 0) ? src : dst;   // reverse conv swaps src/dst
      const int* dD = (d == 0) ? dst : src;
      k_init_dir<<<(Nn + 255) / 256, 256, 0, stream>>>(EMX, DEN, Nn);
      int eb = (Ee + 7) / 8;                  // 8 waves per 256-thread block
      k_edge_logits<<<eb, 256, 0, stream>>>(sD, dD, eattr,
          WeD[d] + (size_t)i * 1024, attD[d] + (size_t)i * 128, XL, XR, EB, EMX, Ee);
      k_edge_exp<<<(Ee + 255) / 256, 256, 0, stream>>>(EB, dD, EMX, DEN, Ee);
      k_edge_scatter<<<eb, 256, 0, stream>>>(sD, dD, EB, DEN, XL, Xn, d * 128, Ee);
    }
    float* t = Xc; Xc = Xn; Xn = t;
  }

  // ------- readout
  k_final<<<(Nn + 7) / 8, 256, 0, stream>>>(Xc, W2, b2, (float*)d_out, Nn);
}
